// NeuralLongTermMemory_1486058684602
// MI455X (gfx1250) — matrix operand order repaired and checked by reading, not compiled, not run on hardware
//
#include <hip/hip_runtime.h>

// ---------------------------------------------------------------------------
// NeuralLongTermMemory for MI455X (gfx1250)
//  K1: fused q/k/v projection GEMMs via v_wmma_f32_16x16x32_bf16
//      (double-buffered LDS staging, fully unrolled vector staging)
//  K2: depthwise conv(K=3) + optional LayerNorm (block reduce)
//  K3: gate MLPs (alpha/theta/eta), batch-averaged per token
//  K4: persistent single-workgroup scan; full optimizer state in 324KB LDS,
//      per-step k/v/q staged with global_load_async_to_lds_b128 (ASYNCcnt)
// ---------------------------------------------------------------------------

typedef __attribute__((ext_vector_type(16))) __bf16 bf16x16;
typedef __attribute__((ext_vector_type(8)))  __bf16 bf16x8;
typedef __attribute__((ext_vector_type(4)))  __bf16 bf16x4;
typedef __attribute__((ext_vector_type(8)))  float  f32x8;
typedef __attribute__((ext_vector_type(4)))  float  f32x4;

union Frag16 { bf16x16 v; bf16x8 h[2]; };

#define S_LEN 1024
#define D_DIM 512
#define M_DIM 512

// ------------------------- K1: projection GEMMs ----------------------------
// out[i,m] = sum_d x[i,d] * W[m,d]; i in [0,8192), m in [0,512), d in [0,512)
// Block: 256 thr (8 waves). Tile 128(i) x 64(m), K-chunk 32, double-buffered.
// Wave (wi,wm) owns 2x2 16x16 WMMA tiles. LDS row pitch 40 bf16 (80B) =>
// 16-row fragment loads land on distinct bank groups; all loads 16B-aligned.
constexpr int BLK_I = 128, BLK_M = 64, BLK_K = 32;
constexpr int XP = 40, WP = 40;

__global__ __launch_bounds__(256)
void proj_wmma_kernel(const float* __restrict__ x,
                      const float* __restrict__ Wk,
                      const float* __restrict__ Wv,
                      const float* __restrict__ Wq,
                      float* __restrict__ kpre,
                      float* __restrict__ vpre,
                      float* __restrict__ qpre)
{
    __shared__ __bf16 xsh[2][BLK_I * XP];
    __shared__ __bf16 wsh[2][BLK_M * WP];

    const float* W; float* out;
    if (blockIdx.z == 0)      { W = Wk; out = kpre; }
    else if (blockIdx.z == 1) { W = Wv; out = vpre; }
    else                      { W = Wq; out = qpre; }

    const int i0   = blockIdx.x * BLK_I;
    const int m0   = blockIdx.y * BLK_M;
    const int tid  = threadIdx.x;
    const int lane = tid & 31;
    const int w    = tid >> 5;
    const int wi   = w & 3;          // i-tile pair index (0..3)
    const int wm   = w >> 2;         // m-tile pair index (0..1)
    const int lhalf = lane >> 4;     // lane group 0/1
    const int lrow  = lane & 15;

    // stage one 32-deep K-chunk into LDS buffer `buf`.
    // Exact, uniform trip counts (4 + 2 iterations of 256 threads) ->
    // fully unrolled, all 6 global_load_b128 issued before first use.
    auto stage = [&](int buf, int k0) {
        f32x4 fx[4], fw[2];
#pragma unroll
        for (int it = 0; it < 4; ++it) {
            int i = tid + it * 256;
            int r = i >> 3, c = (i & 7) * 4;
            fx[it] = *(const f32x4*)&x[(size_t)(i0 + r) * D_DIM + k0 + c];
        }
#pragma unroll
        for (int it = 0; it < 2; ++it) {
            int i = tid + it * 256;
            int r = i >> 3, c = (i & 7) * 4;
            fw[it] = *(const f32x4*)&W[(size_t)(m0 + r) * D_DIM + k0 + c];
        }
#pragma unroll
        for (int it = 0; it < 4; ++it) {
            int i = tid + it * 256;
            int r = i >> 3, c = (i & 7) * 4;
            *(bf16x4*)&xsh[buf][r * XP + c] = __builtin_convertvector(fx[it], bf16x4);
        }
#pragma unroll
        for (int it = 0; it < 2; ++it) {
            int i = tid + it * 256;
            int r = i >> 3, c = (i & 7) * 4;
            *(bf16x4*)&wsh[buf][r * WP + c] = __builtin_convertvector(fw[it], bf16x4);
        }
    };

    f32x8 acc[2][2] = {};
    stage(0, 0);

    constexpr int NK = D_DIM / BLK_K;          // 16
    for (int kk = 0; kk < NK; ++kk) {
        __syncthreads();                        // buf ready; buf^1 free
        if (kk + 1 < NK) stage((kk + 1) & 1, (kk + 1) * BLK_K);

        const __bf16* xb = xsh[kk & 1];
        const __bf16* wb = wsh[kk & 1];

        // A fragments: lane<16 holds K {0..7,16..23}; lane>=16 {8..15,24..31}
        Frag16 a[2], b[2];
#pragma unroll
        for (int ti = 0; ti < 2; ++ti) {
            const __bf16* ap = &xb[((wi * 2 + ti) * 16 + lrow) * XP];
            a[ti].h[0] = *(const bf16x8*)(ap + 8 * lhalf);
            a[ti].h[1] = *(const bf16x8*)(ap + 16 + 8 * lhalf);
        }
        // B fragments: lane<16 holds K {0..7,16..23}; lane>=16 {8..15,24..31}
#pragma unroll
        for (int tj = 0; tj < 2; ++tj) {
            const __bf16* bp = &wb[((wm * 2 + tj) * 16 + lrow) * WP];
            b[tj].h[0] = *(const bf16x8*)(bp + 8 * lhalf);
            b[tj].h[1] = *(const bf16x8*)(bp + 16 + 8 * lhalf);
        }
#pragma unroll
        for (int ti = 0; ti < 2; ++ti)
#pragma unroll
            for (int tj = 0; tj < 2; ++tj)
                acc[ti][tj] = __builtin_amdgcn_wmma_f32_16x16x32_bf16(
                    false, a[ti].v, false, b[tj].v,
                    (short)0, acc[ti][tj], false, false);
    }

    // C/D layout: VGPR r -> (M = r + 8*lhalf, N = lrow)
#pragma unroll
    for (int ti = 0; ti < 2; ++ti)
#pragma unroll
        for (int tj = 0; tj < 2; ++tj)
#pragma unroll
            for (int r = 0; r < 8; ++r) {
                int row = i0 + (wi * 2 + ti) * 16 + r + lhalf * 8;
                int col = m0 + (wm * 2 + tj) * 16 + lrow;
                out[(size_t)row * M_DIM + col] = acc[ti][tj][r];
            }
}

// ---------------------- K2: depthwise conv + LayerNorm ---------------------
__global__ __launch_bounds__(512)
void conv_ln_kernel(const float* __restrict__ kpre, const float* __restrict__ vpre,
                    const float* __restrict__ qpre,
                    const float* __restrict__ ckw, const float* __restrict__ ckb,
                    const float* __restrict__ cvw, const float* __restrict__ cvb,
                    const float* __restrict__ cqw, const float* __restrict__ cqb,
                    const float* __restrict__ lng, const float* __restrict__ lnb,
                    float* __restrict__ ko, float* __restrict__ vo, float* __restrict__ qo)
{
    __shared__ float red[512];
    __shared__ float stat[2];
    const int s = blockIdx.x, b = blockIdx.y, z = blockIdx.z;
    const int m = threadIdx.x;

    const float *pre, *cw, *cb; float* out; bool ln;
    if (z == 0)      { pre = kpre; cw = ckw; cb = ckb; out = ko; ln = true;  }
    else if (z == 1) { pre = vpre; cw = cvw; cb = cvb; out = vo; ln = false; }
    else             { pre = qpre; cw = cqw; cb = cqb; out = qo; ln = true;  }

    const size_t row = ((size_t)b * S_LEN + s) * M_DIM;
    float c = cw[m * 3 + 1] * pre[row + m] + cb[m];
    if (s > 0)         c += cw[m * 3 + 0] * pre[row - M_DIM + m];
    if (s < S_LEN - 1) c += cw[m * 3 + 2] * pre[row + M_DIM + m];

    if (ln) {
        red[m] = c; __syncthreads();
        for (int off = 256; off > 0; off >>= 1) {
            if (m < off) red[m] += red[m + off];
            __syncthreads();
        }
        if (m == 0) stat[0] = red[0] * (1.f / 512.f);
        __syncthreads();
        float mu = stat[0];
        float d = c - mu;
        red[m] = d * d; __syncthreads();
        for (int off = 256; off > 0; off >>= 1) {
            if (m < off) red[m] += red[m + off];
            __syncthreads();
        }
        if (m == 0) stat[1] = red[0] * (1.f / 512.f);
        __syncthreads();
        c = d * rsqrtf(stat[1] + 1e-5f) * lng[m] + lnb[m];
    }
    out[row + m] = c;
}

// --------------------------- K3: gate MLPs ---------------------------------
__global__ __launch_bounds__(256)
void coeff_kernel(const float* __restrict__ x,
                  const float* aW1, const float* ab1, const float* aW2, const float* ab2,
                  const float* tW1, const float* tb1, const float* tW2, const float* tb2,
                  const float* eW1, const float* eb1, const float* eW2, const float* eb2,
                  float* __restrict__ alpha, float* __restrict__ theta,
                  float* __restrict__ eta)
{
    __shared__ float xs[512];
    __shared__ float part[16][17];
    __shared__ float hv[16];
    const int s = blockIdx.x, z = blockIdx.y;
    const float *W1, *b1, *W2, *b2; float* g;
    if (z == 0)      { W1 = aW1; b1 = ab1; W2 = aW2; b2 = ab2; g = alpha; }
    else if (z == 1) { W1 = tW1; b1 = tb1; W2 = tW2; b2 = tb2; g = theta; }
    else             { W1 = eW1; b1 = eb1; W2 = eW2; b2 = eb2; g = eta;   }
    const int tid = threadIdx.x;
    const int h = tid >> 4, sub = tid & 15;
    float csum = 0.f;
    for (int b = 0; b < 8; ++b) {
        __syncthreads();
        for (int i = tid; i < 512; i += 256)
            xs[i] = x[((size_t)b * S_LEN + s) * D_DIM + i];
        __syncthreads();
        float p = 0.f;
        for (int d = sub * 32; d < sub * 32 + 32; ++d) p += xs[d] * W1[h * D_DIM + d];
        part[h][sub] = p;
        __syncthreads();
        if (tid < 16) {
            float t = b1[tid];
            for (int u = 0; u < 16; ++u) t += part[tid][u];
            hv[tid] = t / (1.f + __expf(-t));          // silu
        }
        __syncthreads();
        if (tid == 0) {
            float o = b2[0];
            for (int u = 0; u < 16; ++u) o += W2[u] * hv[u];
            csum += 1.f / (1.f + __expf(-o));          // sigmoid
        }
    }
    __syncthreads();
    if (tid == 0) g[s] = csum * 0.125f;
}

// ----------------- K4: persistent scan (state in 324KB LDS) ----------------
// Pads: W1/S1 rows 513 floats, W2/S2 rows 33 floats -> conflict-free access.
// Per-step k/v/q staging uses the CDNA5 async global->LDS DMA path.
__global__ __launch_bounds__(1024)
void scan_kernel(const float* __restrict__ kA, const float* __restrict__ vA,
                 const float* __restrict__ qA,
                 const float* __restrict__ alpha, const float* __restrict__ theta,
                 const float* __restrict__ eta,
                 const float* __restrict__ W1g, const float* __restrict__ b1g,
                 const float* __restrict__ W2g, const float* __restrict__ b2g,
                 float* __restrict__ out)
{
    extern __shared__ float sm[];
    float* W1s = sm;                   // [32][513]
    float* S1s = W1s + 32 * 513;
    float* W2s = S1s + 32 * 513;       // [512][33]
    float* S2s = W2s + 512 * 33;
    float* b1s = S2s + 512 * 33;       // [32]
    float* Sb1 = b1s + 32;
    float* b2s = Sb1 + 32;             // [512]
    float* Sb2 = b2s + 512;
    float* kts = Sb2 + 512;            // [8][512] (16B-aligned offset)
    float* vts = kts + 4096;
    float* dts = vts + 4096;           // qt during phase1, dout afterwards
    float* hks = dts + 4096;           // [8][32] raw pre-acts (k path)
    float* a1s = hks + 256;
    float* ays = a1s + 256;
    float* dhs = ays + 256;

    const int tid = threadIdx.x;
    for (int i = tid; i < 32 * 512; i += 1024) {
        int j = i >> 9, m = i & 511;
        W1s[j * 513 + m] = W1g[i]; S1s[j * 513 + m] = 0.f;
    }
    for (int i = tid; i < 512 * 32; i += 1024) {
        int m = i >> 5, j = i & 31;
        W2s[m * 33 + j] = W2g[i]; S2s[m * 33 + j] = 0.f;
    }
    if (tid < 32)  { b1s[tid] = b1g[tid]; Sb1[tid] = 0.f; }
    if (tid < 512) { b2s[tid] = b2g[tid]; Sb2[tid] = 0.f; }
    __syncthreads();

    const float inv = 2.f / 4096.f;   // d(mean sq err)/d(out)

    // per-thread 16B quad for async staging: 4096 floats = 1024 quads/array
    const int sb = tid >> 7;          // batch row 0..7
    const int smm = (tid & 127) * 4;  // column (multiple of 4 floats = 16B)
    const uint32_t lds_k = (uint32_t)(size_t)(void*)&kts[sb * 512 + smm];
    const uint32_t lds_v = (uint32_t)(size_t)(void*)&vts[sb * 512 + smm];
    const uint32_t lds_q = (uint32_t)(size_t)(void*)&dts[sb * 512 + smm];

    for (int t = 0; t < S_LEN; ++t) {
        // ---- stage k_t, v_t, q_t via async global->LDS DMA (ASYNCcnt) ----
        {
            size_t off = ((size_t)sb * S_LEN + t) * M_DIM + smm;
            asm volatile("global_load_async_to_lds_b128 %0, %1, off"
                         :: "v"(lds_k), "v"(kA + off) : "memory");
            asm volatile("global_load_async_to_lds_b128 %0, %1, off"
                         :: "v"(lds_v), "v"(vA + off) : "memory");
            asm volatile("global_load_async_to_lds_b128 %0, %1, off"
                         :: "v"(lds_q), "v"(qA + off) : "memory");
            if ((smm & 15) == 0 && t + 1 < S_LEN) {   // 1 prefetch / 64B line
                __builtin_prefetch(kA + off + M_DIM, 0, 0);
                __builtin_prefetch(vA + off + M_DIM, 0, 0);
                __builtin_prefetch(qA + off + M_DIM, 0, 0);
            }
            asm volatile("s_wait_asynccnt 0x0" ::: "memory");
        }
        __syncthreads();

        // phase 1: h = in @ W1.T + b1 (512 length-512 dots: k-path + q-path)
        if (tid < 512) {
            int which = tid >> 8, b = (tid >> 5) & 7, j = tid & 31;
            const float* irow = (which ? dts : kts) + b * 512;
            const float* wrow = &W1s[j * 513];
            float acc = b1s[j];
            for (int m = 0; m < 512; ++m) acc += irow[m] * wrow[m];
            float sg = 1.f / (1.f + __expf(-acc));
            if (which == 0) { hks[b * 32 + j] = acc; a1s[b * 32 + j] = acc * sg; }
            else            { ays[b * 32 + j] = acc * sg; }
        }
        __syncthreads();

        // phase 2: out = a @ W2.T + b2 for both k-loss (-> dout) and y (-> global)
#pragma unroll
        for (int r = 0; r < 8; ++r) {
            int idx = tid + r * 1024;
            int which = idx >> 12, b = (idx >> 9) & 7, m = idx & 511;
            const float* arow = (which ? ays : a1s) + b * 32;
            const float* wrow = &W2s[m * 33];
            float acc = b2s[m];
#pragma unroll
            for (int j = 0; j < 32; ++j) acc += arow[j] * wrow[j];
            if (which == 0) dts[b * 512 + m] = (acc - vts[b * 512 + m]) * inv;
            else            out[((size_t)b * S_LEN + t) * M_DIM + m] = acc;
        }
        __syncthreads();

        // phase 3: da = dout @ W2; dh = da * silu'(h)
        if (tid < 256) {
            int b = tid >> 5, j = tid & 31;
            const float* drow = &dts[b * 512];
            float acc = 0.f;
            for (int m = 0; m < 512; ++m) acc += drow[m] * W2s[m * 33 + j];
            float h = hks[b * 32 + j];
            float sg = 1.f / (1.f + __expf(-h));
            dhs[b * 32 + j] = acc * sg * (1.f + h * (1.f - sg));
        }
        __syncthreads();

        // phase 4: rank-8 gradient outer products fused with momentum update
        float al = alpha[t], th = theta[t], et = eta[t];
        float oma = 1.f - al;
#pragma unroll
        for (int r = 0; r < 16; ++r) {
            int idx = tid + r * 1024;
            int m = idx >> 5, j = idx & 31;
            float gsum = 0.f;
#pragma unroll
            for (int b = 0; b < 8; ++b) gsum += dts[b * 512 + m] * a1s[b * 32 + j];
            float sv = et * S2s[m * 33 + j] - th * gsum;
            S2s[m * 33 + j] = sv;
            W2s[m * 33 + j] = oma * W2s[m * 33 + j] + sv;
        }
#pragma unroll
        for (int r = 0; r < 16; ++r) {
            int idx = tid + r * 1024;
            int j = idx >> 9, m = idx & 511;
            float gsum = 0.f;
#pragma unroll
            for (int b = 0; b < 8; ++b) gsum += dhs[b * 32 + j] * kts[b * 512 + m];
            float sv = et * S1s[j * 513 + m] - th * gsum;
            S1s[j * 513 + m] = sv;
            W1s[j * 513 + m] = oma * W1s[j * 513 + m] + sv;
        }
        if (tid < 512) {
            float gsum = 0.f;
            for (int b = 0; b < 8; ++b) gsum += dts[b * 512 + tid];
            float sv = et * Sb2[tid] - th * gsum;
            Sb2[tid] = sv; b2s[tid] = oma * b2s[tid] + sv;
        }
        if (tid < 32) {
            float gsum = 0.f;
            for (int b = 0; b < 8; ++b) gsum += dhs[b * 32 + tid];
            float sv = et * Sb1[tid] - th * gsum;
            Sb1[tid] = sv; b1s[tid] = oma * b1s[tid] + sv;
        }
        __syncthreads();
    }
}

// ------------------------------- launcher ----------------------------------
extern "C" void kernel_launch(void* const* d_in, const int* in_sizes, int n_in,
                              void* d_out, int out_size, void* d_ws, size_t ws_size,
                              hipStream_t stream)
{
    const float* x   = (const float*)d_in[0];
    const float* Wk  = (const float*)d_in[1];
    const float* Wv  = (const float*)d_in[2];
    const float* Wq  = (const float*)d_in[3];
    const float* ckw = (const float*)d_in[4];
    const float* ckb = (const float*)d_in[5];
    const float* cvw = (const float*)d_in[6];
    const float* cvb = (const float*)d_in[7];
    const float* cqw = (const float*)d_in[8];
    const float* cqb = (const float*)d_in[9];
    const float* lng = (const float*)d_in[10];
    const float* lnb = (const float*)d_in[11];
    const float* W1  = (const float*)d_in[12];
    const float* b1  = (const float*)d_in[13];
    const float* W2  = (const float*)d_in[14];
    const float* b2  = (const float*)d_in[15];
    const float* aW1 = (const float*)d_in[16];
    const float* ab1 = (const float*)d_in[17];
    const float* aW2 = (const float*)d_in[18];
    const float* ab2 = (const float*)d_in[19];
    const float* tW1 = (const float*)d_in[20];
    const float* tb1 = (const float*)d_in[21];
    const float* tW2 = (const float*)d_in[22];
    const float* tb2 = (const float*)d_in[23];
    const float* eW1 = (const float*)d_in[24];
    const float* eb1 = (const float*)d_in[25];
    const float* eW2 = (const float*)d_in[26];
    const float* eb2 = (const float*)d_in[27];

    float* ws = (float*)d_ws;
    const size_t NBSM = (size_t)8 * S_LEN * M_DIM;
    float* kpre = ws;
    float* vpre = kpre + NBSM;
    float* qpre = vpre + NBSM;
    float* kk   = qpre + NBSM;
    float* vv   = kk + NBSM;
    float* qq   = vv + NBSM;
    float* alpha = qq + NBSM;
    float* theta = alpha + S_LEN;
    float* eta   = theta + S_LEN;
    float* out = (float*)d_out;

    proj_wmma_kernel<<<dim3(8192 / BLK_I, M_DIM / BLK_M, 3), 256, 0, stream>>>(
        x, Wk, Wv, Wq, kpre, vpre, qpre);
    conv_ln_kernel<<<dim3(S_LEN, 8, 3), 512, 0, stream>>>(
        kpre, vpre, qpre, ckw, ckb, cvw, cvb, cqw, cqb, lng, lnb, kk, vv, qq);
    coeff_kernel<<<dim3(S_LEN, 3), 256, 0, stream>>>(
        x, aW1, ab1, aW2, ab2, tW1, tb1, tW2, tb2, eW1, eb1, eW2, eb2,
        alpha, theta, eta);

    size_t smem = (size_t)(2 * 32 * 513 + 2 * 512 * 33 + 2 * 32 + 2 * 512 +
                           3 * 4096 + 4 * 256) * sizeof(float);   // 324,096 B
    scan_kernel<<<1, 1024, smem, stream>>>(
        kk, vv, qq, alpha, theta, eta, W1, b1, W2, b2, out);
}
